// StreamingPartialUntiedAssociativeLM_69690139344925
// MI455X (gfx1250) — compile-verified, matrix operand order checked by reading, not run on hardware
//
#include <hip/hip_runtime.h>
#include <math.h>

namespace {

constexpr int B = 4, S = 512, V = 32000, E = 512, H = 1024, DM = 256;
constexpr int U = 4096, M = 1024;
constexpr int MS = M + S;      // 1536  (< MEM_CAP=2048 -> full concat kept)
constexpr int BS = B * S;      // 2048 GEMM rows

typedef __attribute__((ext_vector_type(16))) __bf16 v16bf;
typedef __attribute__((ext_vector_type(8)))  __bf16 v8bf;
typedef __attribute__((ext_vector_type(8)))  float  v8f;

__device__ __forceinline__ unsigned short f2bf(float f) {
  union { float f; unsigned u; } v; v.f = f;
  unsigned r = v.u + 0x7FFFu + ((v.u >> 16) & 1u);   // round-to-nearest-even
  return (unsigned short)(r >> 16);
}

// ---------------------------------------------------------------- converts
__global__ void cvt_bf16_kernel(const float* __restrict__ src,
                                unsigned short* __restrict__ dst, int n) {
  int i = blockIdx.x * blockDim.x + threadIdx.x;
  if (i < n) dst[i] = f2bf(src[i]);
}

__global__ void embed_kernel(const int* __restrict__ ids,
                             const float* __restrict__ emb,
                             unsigned short* __restrict__ xbf) {
  int row = blockIdx.x;                           // [0, B*S)
  const float* src = emb + (size_t)ids[row] * E;
  unsigned short* dst = xbf + (size_t)row * E;
  for (int i = threadIdx.x; i < E; i += blockDim.x) dst[i] = f2bf(src[i]);
}

// ------------------------------------------------------- WMMA NT GEMM
// D[m,n] = act( scale * sum_k A[m,k]*W[n,k] + bias[n] )
// A: [Mdim,K] bf16 row-major; W: [Ndim,K] bf16 row-major (K contiguous both).
// Per wave: 32x32 output (2x2 WMMA tiles, fragments reused twice each).
// Block = 8 waves (2x4) = 64x128 tile. All dims: M%64==0, N%128==0, K%32==0.
template <int ACT, int WF32, int WBF16>
__global__ __launch_bounds__(256) void wmma_gemm_nt(
    const unsigned short* __restrict__ A, long long sA,
    const unsigned short* __restrict__ W, long long sW,
    const float* __restrict__ bias, float scale,
    float* __restrict__ D, long long sD,
    unsigned short* __restrict__ Dbf, long long sDbf,
    int Ndim, int Kdim) {
  const int wave = threadIdx.x >> 5;
  const int lane = threadIdx.x & 31;
  const int half = lane >> 4;      // lanes 16..31 hold the upper-K halves
  const int r16  = lane & 15;
  const int m0 = blockIdx.y * 64 + (wave >> 2) * 32;
  const int n0 = blockIdx.x * 128 + (wave & 3) * 32;

  const unsigned short* Ab = A + (size_t)blockIdx.z * sA;
  const unsigned short* Wb = W + (size_t)blockIdx.z * sW;
  const unsigned short* ap0 = Ab + (size_t)(m0 + r16) * Kdim + half * 8;
  const unsigned short* ap1 = ap0 + (size_t)16 * Kdim;
  const unsigned short* wp0 = Wb + (size_t)(n0 + r16) * Kdim + half * 8;
  const unsigned short* wp1 = wp0 + (size_t)16 * Kdim;

  v8f acc00 = {}, acc01 = {}, acc10 = {}, acc11 = {};
  for (int k = 0; k < Kdim; k += 32) {
    __builtin_prefetch(ap0 + k + 128, 0, 1);      // global_prefetch_b8
    __builtin_prefetch(ap1 + k + 128, 0, 1);
    __builtin_prefetch(wp0 + k + 128, 0, 1);
    __builtin_prefetch(wp1 + k + 128, 0, 1);
    union { v16bf v; v8bf h[2]; } a0, a1, w0, w1;
    a0.h[0] = *(const v8bf*)(ap0 + k);  a0.h[1] = *(const v8bf*)(ap0 + k + 16);
    a1.h[0] = *(const v8bf*)(ap1 + k);  a1.h[1] = *(const v8bf*)(ap1 + k + 16);
    w0.h[0] = *(const v8bf*)(wp0 + k);  w0.h[1] = *(const v8bf*)(wp0 + k + 16);
    w1.h[0] = *(const v8bf*)(wp1 + k);  w1.h[1] = *(const v8bf*)(wp1 + k + 16);
    acc00 = __builtin_amdgcn_wmma_f32_16x16x32_bf16(false, a0.v, false, w0.v,
                                                    (short)0, acc00, false, false);
    acc01 = __builtin_amdgcn_wmma_f32_16x16x32_bf16(false, a0.v, false, w1.v,
                                                    (short)0, acc01, false, false);
    acc10 = __builtin_amdgcn_wmma_f32_16x16x32_bf16(false, a1.v, false, w0.v,
                                                    (short)0, acc10, false, false);
    acc11 = __builtin_amdgcn_wmma_f32_16x16x32_bf16(false, a1.v, false, w1.v,
                                                    (short)0, acc11, false, false);
  }

  float* Db = D + (size_t)blockIdx.z * sD;
  unsigned short* Dbfb = Dbf + (size_t)blockIdx.z * sDbf;
  const v8f accA[2][2] = {{acc00, acc01}, {acc10, acc11}};
  const float bvs[2] = {bias ? bias[n0 + r16] : 0.0f,
                        bias ? bias[n0 + 16 + r16] : 0.0f};
#pragma unroll
  for (int mi = 0; mi < 2; ++mi) {
#pragma unroll
    for (int ni = 0; ni < 2; ++ni) {
#pragma unroll
      for (int r = 0; r < 8; ++r) {
        const int mrow = m0 + mi * 16 + r + half * 8;   // lanes16-31 -> M+8
        const int col  = n0 + ni * 16 + r16;
        float vout = accA[mi][ni][r] * scale + bvs[ni];
        if (ACT == 1) { float t = fmaxf(vout, 0.f); vout = t * t; } // relu^2
        const size_t o = (size_t)mrow * Ndim + col;
        if (WF32)  __builtin_nontemporal_store(vout, Db + o);  // streamed once
        if (WBF16) Dbfb[o] = f2bf(vout);                       // re-read: keep RT
      }
    }
  }
}

// ------------------------------------------------------------- GRU scan
// grid = B blocks, 1024 threads; thread t owns hidden unit t; h lives in LDS.
__global__ __launch_bounds__(1024) void gru_scan_kernel(
    const float* __restrict__ xp,      // [B,S,3H]
    const float* __restrict__ whh,     // [3H,H]
    const float* __restrict__ bhh,     // [3H]
    float* __restrict__ states,        // [B,S,H]
    unsigned short* __restrict__ states_bf) {
  __shared__ float hsh[H];
  const int b = blockIdx.x;
  const int t = threadIdx.x;
  hsh[t] = 0.0f;
  __syncthreads();

  const float4* w0 = (const float4*)(whh + (size_t)t * H);
  const float4* w1 = (const float4*)(whh + (size_t)(H + t) * H);
  const float4* w2 = (const float4*)(whh + (size_t)(2 * H + t) * H);
  const float b0 = bhh[t], b1 = bhh[H + t], b2 = bhh[2 * H + t];

  for (int s = 0; s < S; ++s) {
    const float* xpt = xp + ((size_t)(b * S + s)) * (3 * H);
    float a0 = b0, a1 = b1, a2 = b2;
    const float4* hv = (const float4*)hsh;
#pragma unroll 4
    for (int k = 0; k < H / 4; ++k) {
      float4 hh = hv[k];
      float4 c0 = w0[k], c1 = w1[k], c2 = w2[k];
      a0 += hh.x * c0.x + hh.y * c0.y + hh.z * c0.z + hh.w * c0.w;
      a1 += hh.x * c1.x + hh.y * c1.y + hh.z * c1.z + hh.w * c1.w;
      a2 += hh.x * c2.x + hh.y * c2.y + hh.z * c2.z + hh.w * c2.w;
    }
    const float hold = hsh[t];
    const float r = 1.f / (1.f + expf(-(xpt[t] + a0)));
    const float z = 1.f / (1.f + expf(-(xpt[H + t] + a1)));
    const float nn = tanhf(xpt[2 * H + t] + r * a2);
    const float hn = (1.f - z) * nn + z * hold;
    __syncthreads();          // everyone done reading old h
    hsh[t] = hn;
    const size_t off = (size_t)(b * S + s) * H + t;
    states[off] = hn;
    states_bf[off] = f2bf(hn);
    __syncthreads();          // h updated for next step
  }
}

// ------------------------------------------------------------- gate
__global__ void gate_kernel(const float* __restrict__ states,
                            const float* __restrict__ gw,
                            const float* __restrict__ gb,
                            float* __restrict__ gate) {
  const int row = blockIdx.x * (blockDim.x >> 5) + (threadIdx.x >> 5);
  const int lane = threadIdx.x & 31;
  if (row >= BS) return;
  const float* sp = states + (size_t)row * H;
  float acc = 0.f;
  for (int k = lane; k < H; k += 32) acc += sp[k] * gw[k];
  for (int o = 16; o; o >>= 1) acc += __shfl_xor(acc, o, 32);
  if (lane == 0) gate[row] = 1.f / (1.f + expf(-(acc + gb[0])));
}

// -------------------------------------------- softmax + token scatter-add
__global__ __launch_bounds__(256) void softmax_scatter_kernel(
    const float* __restrict__ scores,       // [B,S,MS], already scaled
    const float* __restrict__ gate,         // [B*S]
    const float* __restrict__ mem_scale,    // [1]
    const int* __restrict__ past_tokens,    // [B,M]
    const int* __restrict__ input_ids,      // [B,S]
    float* __restrict__ logits) {           // [B*S,V]
  __shared__ float red[256];
  const int bs = blockIdx.x;
  const int b = bs / S, s = bs % S;
  const int tid = threadIdx.x;
  const float* row = scores + (size_t)bs * MS;
  const int L = M + s;                      // valid: all past + strictly-before cur

  float mx = -3.4e38f;
  for (int t = tid; t < L; t += 256) mx = fmaxf(mx, row[t]);
  red[tid] = mx; __syncthreads();
  for (int o = 128; o; o >>= 1) { if (tid < o) red[tid] = fmaxf(red[tid], red[tid + o]); __syncthreads(); }
  mx = red[0]; __syncthreads();

  float sum = 0.f;
  for (int t = tid; t < L; t += 256) sum += expf(row[t] - mx);
  red[tid] = sum; __syncthreads();
  for (int o = 128; o; o >>= 1) { if (tid < o) red[tid] += red[tid + o]; __syncthreads(); }
  const float g = gate[bs] * mem_scale[0] / fmaxf(red[0], 1e-6f);

  float* lrow = logits + (size_t)bs * V;
  for (int t = tid; t < L; t += 256) {
    const float w = expf(row[t] - mx) * g;
    const int tok = (t < M) ? past_tokens[b * M + t] : input_ids[b * S + (t - M)];
    atomicAdd(lrow + tok, w);
  }
}

// ---------------------------------------------------- untied partial scatter
__global__ void scatter_partial_kernel(const float* __restrict__ partial, // [B*S,U]
                                       const int* __restrict__ untied,    // [U]
                                       float* __restrict__ logits) {
  const size_t idx = (size_t)blockIdx.x * blockDim.x + threadIdx.x;
  if (idx >= (size_t)BS * U) return;
  const int u = (int)(idx & (U - 1));     // U = 4096 = 2^12
  const size_t row = idx >> 12;
  atomicAdd(logits + row * V + untied[u], partial[idx]);
}

// --------------------------------------- assemble keys/tokens outputs
__global__ void assemble_keys_kernel(const float* __restrict__ past_keys, // [B,M,DM]
                                     const float* __restrict__ ck,        // [B,S,DM]
                                     unsigned short* __restrict__ keys_bf,// [B,MS,DM]
                                     float* __restrict__ out_keys) {      // [B,MS,DM]
  const size_t idx = (size_t)blockIdx.x * blockDim.x + threadIdx.x;
  if (idx >= (size_t)B * MS * DM) return;
  const int d = (int)(idx % DM);
  const int m = (int)((idx / DM) % MS);
  const int b = (int)(idx / ((size_t)MS * DM));
  const float v = (m < M) ? past_keys[((size_t)b * M + m) * DM + d]
                          : ck[((size_t)b * S + (m - M)) * DM + d];
  keys_bf[idx] = f2bf(v);
  out_keys[idx] = v;
}

__global__ void assemble_tokens_kernel(const int* __restrict__ past_tokens,
                                       const int* __restrict__ input_ids,
                                       int* __restrict__ out_tokens) {    // [B,MS]
  const int idx = blockIdx.x * blockDim.x + threadIdx.x;
  if (idx >= B * MS) return;
  const int m = idx % MS, b = idx / MS;
  out_tokens[idx] = (m < M) ? past_tokens[b * M + m] : input_ids[b * S + (m - M)];
}

}  // namespace

// =========================================================== host launcher
extern "C" void kernel_launch(void* const* d_in, const int* in_sizes, int n_in,
                              void* d_out, int out_size, void* d_ws, size_t ws_size,
                              hipStream_t stream) {
  (void)in_sizes; (void)n_in; (void)out_size; (void)ws_size;

  const int*   input_ids   = (const int*)  d_in[0];
  const int*   past_tokens = (const int*)  d_in[1];
  const float* past_keys   = (const float*)d_in[2];
  const int*   untied_ids  = (const int*)  d_in[3];
  const float* emb         = (const float*)d_in[4];
  const float* gru_w_ih    = (const float*)d_in[5];
  const float* gru_w_hh    = (const float*)d_in[6];
  const float* gru_b_ih    = (const float*)d_in[7];
  const float* gru_b_hh    = (const float*)d_in[8];
  const float* q_w         = (const float*)d_in[9];
  const float* q_b         = (const float*)d_in[10];
  const float* k_w         = (const float*)d_in[11];
  const float* k_b         = (const float*)d_in[12];
  const float* gate_w      = (const float*)d_in[13];
  const float* gate_b      = (const float*)d_in[14];
  const float* fc_w        = (const float*)d_in[15];
  const float* fc_b        = (const float*)d_in[16];
  const float* proj_w      = (const float*)d_in[17];
  const float* proj_b      = (const float*)d_in[18];
  const float* out_bias    = (const float*)d_in[19];
  const float* part_w      = (const float*)d_in[20];
  const float* part_b      = (const float*)d_in[21];
  const float* mem_scale   = (const float*)d_in[22];

  float* logits     = (float*)d_out;                                   // [B*S,V]
  float* out_keys   = logits + (size_t)BS * V;                         // [B,MS,DM]
  int*   out_tokens = (int*)(out_keys + (size_t)B * MS * DM);          // [B,MS]

  // ---- workspace carve-up (256B aligned)
  char* base = (char*)d_ws;
  size_t off = 0;
  auto take = [&](size_t bytes) -> char* {
    char* p = base + off;
    off += (bytes + 255) & ~(size_t)255;
    return p;
  };
  unsigned short* emb_bf    = (unsigned short*)take((size_t)V * E * 2);
  unsigned short* wih_bf    = (unsigned short*)take((size_t)3 * H * E * 2);
  unsigned short* fcw_bf    = (unsigned short*)take((size_t)4 * E * H * 2);
  unsigned short* projw_bf  = (unsigned short*)take((size_t)E * 4 * E * 2);
  unsigned short* partw_bf  = (unsigned short*)take((size_t)U * E * 2);
  unsigned short* qw_bf     = (unsigned short*)take((size_t)DM * H * 2);
  unsigned short* kw_bf     = (unsigned short*)take((size_t)DM * H * 2);
  unsigned short* x_bf      = (unsigned short*)take((size_t)BS * E * 2);
  float*          xp        = (float*)        take((size_t)BS * 3 * H * 4);
  float*          states    = (float*)        take((size_t)BS * H * 4);
  unsigned short* states_bf = (unsigned short*)take((size_t)BS * H * 2);
  unsigned short* head_bf   = (unsigned short*)take((size_t)BS * 4 * E * 2);
  unsigned short* base_bf   = (unsigned short*)take((size_t)BS * E * 2);
  float*          partial   = (float*)        take((size_t)BS * U * 4);
  unsigned short* q_bf      = (unsigned short*)take((size_t)BS * DM * 2);
  float*          ckf       = (float*)        take((size_t)BS * DM * 4);
  unsigned short* keys_bf   = (unsigned short*)take((size_t)B * MS * DM * 2);
  float*          scores    = (float*)        take((size_t)BS * MS * 4);
  float*          gate      = (float*)        take((size_t)BS * 4);

  auto cvt = [&](const float* s, unsigned short* d, int n) {
    cvt_bf16_kernel<<<(n + 255) / 256, 256, 0, stream>>>(s, d, n);
  };
  // 1) weight conversions to bf16
  cvt(emb,      emb_bf,   V * E);
  cvt(gru_w_ih, wih_bf,   3 * H * E);
  cvt(fc_w,     fcw_bf,   4 * E * H);
  cvt(proj_w,   projw_bf, E * 4 * E);
  cvt(part_w,   partw_bf, U * E);
  cvt(q_w,      qw_bf,    DM * H);
  cvt(k_w,      kw_bf,    DM * H);

  // 2) embedding gather (bf16)
  embed_kernel<<<BS, 128, 0, stream>>>(input_ids, emb, x_bf);

  // 3) xp = x @ w_ih^T + b_ih          [2048 x 3072, K=512]
  wmma_gemm_nt<0, 1, 0><<<dim3(3 * H / 128, BS / 64, 1), 256, 0, stream>>>(
      x_bf, 0, wih_bf, 0, gru_b_ih, 1.0f, xp, 0, nullptr, 0, 3 * H, E);

  // 4) sequential GRU scan -> states (f32 + bf16)
  gru_scan_kernel<<<B, H, 0, stream>>>(xp, gru_w_hh, gru_b_hh, states, states_bf);

  // 5) head = relu(states @ fc_w^T + fc_b)^2 -> bf16   [2048 x 2048, K=1024]
  wmma_gemm_nt<1, 0, 1><<<dim3(4 * E / 128, BS / 64, 1), 256, 0, stream>>>(
      states_bf, 0, fcw_bf, 0, fc_b, 1.0f, nullptr, 0, head_bf, 0, 4 * E, H);

  // 6) base = head @ proj_w^T + proj_b -> bf16          [2048 x 512, K=2048]
  wmma_gemm_nt<0, 0, 1><<<dim3(E / 128, BS / 64, 1), 256, 0, stream>>>(
      head_bf, 0, projw_bf, 0, proj_b, 1.0f, nullptr, 0, base_bf, 0, E, 4 * E);

  // 7) logits = base @ emb^T + out_bias -> d_out        [2048 x 32000, K=512]
  wmma_gemm_nt<0, 1, 0><<<dim3(V / 128, BS / 64, 1), 256, 0, stream>>>(
      base_bf, 0, emb_bf, 0, out_bias, 1.0f, logits, 0, nullptr, 0, V, E);

  // 8) partial = base @ part_w^T + part_b               [2048 x 4096, K=512]
  wmma_gemm_nt<0, 1, 0><<<dim3(U / 128, BS / 64, 1), 256, 0, stream>>>(
      base_bf, 0, partw_bf, 0, part_b, 1.0f, partial, 0, nullptr, 0, U, E);

  // 9) logits[:, untied_ids] += partial   (after logits GEMM)
  scatter_partial_kernel<<<(int)(((size_t)BS * U + 255) / 256), 256, 0, stream>>>(
      partial, untied_ids, logits);

  // 10) q = states @ q_w^T + q_b -> bf16                 [2048 x 256, K=1024]
  wmma_gemm_nt<0, 0, 1><<<dim3(DM / 128, BS / 64, 1), 256, 0, stream>>>(
      states_bf, 0, qw_bf, 0, q_b, 1.0f, nullptr, 0, q_bf, 0, DM, H);

  // 11) ck = states @ k_w^T + k_b (f32, feeds next_keys + keys_bf)
  wmma_gemm_nt<0, 1, 0><<<dim3(DM / 128, BS / 64, 1), 256, 0, stream>>>(
      states_bf, 0, kw_bf, 0, k_b, 1.0f, ckf, 0, nullptr, 0, DM, H);

  // 12/13) assemble keys (bf16 for scores + f32 out) and tokens out
  assemble_keys_kernel<<<(int)(((size_t)B * MS * DM + 255) / 256), 256, 0, stream>>>(
      past_keys, ckf, keys_bf, out_keys);
  assemble_tokens_kernel<<<(B * MS + 255) / 256, 256, 0, stream>>>(
      past_tokens, input_ids, out_tokens);

  // 14) scores = q @ keys_all^T * (1/sqrt(DM))  batched over B  [512 x 1536, K=256]
  wmma_gemm_nt<0, 1, 0><<<dim3(MS / 128, S / 64, B), 256, 0, stream>>>(
      q_bf, (long long)S * DM, keys_bf, (long long)MS * DM, nullptr, 0.0625f,
      scores, (long long)S * MS, nullptr, 0, MS, DM);

  // 15) gate = sigmoid(states @ gate_w^T + gate_b)
  gate_kernel<<<BS / 8, 256, 0, stream>>>(states, gate_w, gate_b, gate);

  // 16) softmax over [past | causal cur], gate*mem_scale, scatter into logits
  softmax_scatter_kernel<<<BS, 256, 0, stream>>>(
      scores, gate, mem_scale, past_tokens, input_ids, logits);
}
